// Subgraphnet_42606075576555
// MI455X (gfx1250) — matrix operand order repaired and checked by reading, not compiled, not run on hardware
//
#include <hip/hip_runtime.h>

typedef __attribute__((ext_vector_type(16))) _Float16 v16h;
typedef __attribute__((ext_vector_type(8)))  _Float16 v8h;
typedef __attribute__((ext_vector_type(8)))  float    v8f;

__device__ __forceinline__ v8f wmma_f32_f16(v16h a, v16h b, v8f c) {
  // D = A(16x32 f16) * B(32x16 f16) + C(16x16 f32)
  return __builtin_amdgcn_wmma_f32_16x16x32_f16(false, a, false, b, (short)0, c, false, false);
}
// Fast silu: v_exp_f32 + v_rcp_f32 (activations are stored as f16 anyway;
// the IEEE-correct division sequence is ~10 VALU ops vs 1 here).
__device__ __forceinline__ float silu_f(float x) {
  return x * __builtin_amdgcn_rcpf(1.f + __expf(-x));
}

// Wave-private LDS slices + same-wave DS ordering only need a compiler
// scheduling fence, not a workgroup barrier (CDNA5 LDS ops from one wave are
// processed in order).
__device__ __forceinline__ void wave_fence() { __builtin_amdgcn_wave_barrier(); }

// Build a 16x32 f16 A-fragment from an LDS-resident row-major tile.
// Layout (ISA 7.12.2): lane<16 -> M=lane, halves[0..7]=K(kt*32+0..7), halves[8..15]=K(+16..23)
//                      lane>=16 -> M=lane-16, halves = K(+8..15), K(+24..31)
__device__ __forceinline__ v16h lds_afrag(const _Float16* base, int pitch, int kt) {
  const int lane = threadIdx.x & 31;
  const _Float16* p = base + (lane & 15) * pitch + kt * 32 + ((lane >> 4) << 3);
  v8h lo = *(const v8h*)p;
  v8h hi = *(const v8h*)(p + 16);
  return __builtin_shufflevector(lo, hi, 0,1,2,3,4,5,6,7,8,9,10,11,12,13,14,15);
}

// B-fragment: 16 contiguous halves per lane (fragment-major packed weights).
__device__ __forceinline__ v16h lds_bfrag(const _Float16* w, int frag) {
  const int lane = threadIdx.x & 31;
  const _Float16* p = w + (frag * 32 + lane) * 16;
  v8h lo = *(const v8h*)p;
  v8h hi = *(const v8h*)(p + 8);
  return __builtin_shufflevector(lo, hi, 0,1,2,3,4,5,6,7,8,9,10,11,12,13,14,15);
}

// Store a 16x16 f32 C/D tile (lane layout: col = lane%16, rows = v + 8*(lane/16))
// as silu(acc+bias) in f16 into an LDS activation tile.
__device__ __forceinline__ void store_act_silu(_Float16* base, int pitch, int nt, v8f acc, float bias) {
  const int lane = threadIdx.x & 31;
  const int col = nt * 16 + (lane & 15);
  const int r0 = (lane >> 4) << 3;
#pragma unroll
  for (int v = 0; v < 8; ++v)
    base[(r0 + v) * pitch + col] = (_Float16)silu_f(acc[v] + bias);
}

// ---------------------------------------------------------------------------
// Weight pre-swizzle: row-major f32 W[K0][N0] -> fragment-major f16, zero-padded.
// B-fragment (32x16 f16): lane<16 -> col=nt*16+lane, halves[0..7]=K(kt*32+0..7), halves[8..15]=K(+16..23);
//                         lane>=16 -> col=nt*16+lane-16, halves = K(+8..15), K(+24..31)
// dst index ((nt*KT+kt)*32+lane)*16+q == flat thread id t.
// ---------------------------------------------------------------------------
__global__ void prep_w(const float* __restrict__ src, _Float16* __restrict__ dst,
                       int K0, int N0, int KT, int NT) {
  int t = blockIdx.x * blockDim.x + threadIdx.x;
  int total = KT * NT * 512;
  if (t >= total) return;
  int q = t & 15, lane = (t >> 4) & 31, rest = t >> 9;
  int kt = rest % KT, nt = rest / KT;
  int col = nt * 16 + (lane & 15);
  int k = kt * 32 + ((q >> 3) << 4) + ((lane >> 4) << 3) + (q & 7);
  float v = (k < K0 && col < N0) ? src[k * N0 + col] : 0.f;
  dst[t] = (_Float16)v;
}

__global__ void zero_f32(float* p, int n) {
  int t = blockIdx.x * blockDim.x + threadIdx.x;
  if (t < n) p[t] = 0.f;
}

// ---- ug2 = (ug @ ug != 0) via 512-bit mask intersection ---------------------
__global__ void pack_masks(const float* __restrict__ edge, unsigned* __restrict__ rowm,
                           unsigned* __restrict__ colm, int N) {
  int t = blockIdx.x * blockDim.x + threadIdx.x;
  int words = N / 32;
  if (t >= N * words) return;
  int r = t / words, w = t % words;
  unsigned rb = 0, cb = 0;
  for (int b = 0; b < 32; ++b) {
    if (edge[r * N + w * 32 + b] != 0.f) rb |= 1u << b;   // rowmask[r] bit k
    if (edge[(w * 32 + b) * N + r] != 0.f) cb |= 1u << b; // colmask[c=r] bit k
  }
  rowm[t] = rb; colm[t] = cb;
}

__global__ void ug2_kernel(const unsigned* __restrict__ rowm, const unsigned* __restrict__ colm,
                           unsigned char* __restrict__ ug2, int N) {
  int t = blockIdx.x * blockDim.x + threadIdx.x;
  if (t >= N * N) return;
  int a = t / N, b = t % N, words = N / 32;
  unsigned any = 0;
  for (int w = 0; w < words; ++w) any |= rowm[a * words + w] & colm[b * words + w];
  ug2[t] = any ? 1 : 0;
}

// ---- per-scale top-k: sigmoid scores + bitonic sort (N=512, one WG/scale) ---
__global__ void __launch_bounds__(256) topk_kernel(const float* __restrict__ feat,
                                                   const float* __restrict__ wp,
                                                   const float* __restrict__ bp,
                                                   int* __restrict__ idxOut, float* __restrict__ valOut,
                                                   int N, int k0, int k1, int k2) {
  __shared__ float sk[512];
  __shared__ int si[512];
  const int s = blockIdx.x, tid = threadIdx.x;
  for (int r = tid; r < N; r += 256) {
    float acc = bp[s];
    for (int d = 0; d < 64; ++d) acc += feat[r * 64 + d] * wp[s * 64 + d];
    sk[r] = 1.f / (1.f + __expf(-acc));
    si[r] = r;
  }
  __syncthreads();
  for (int k = 2; k <= N; k <<= 1)
    for (int j = k >> 1; j > 0; j >>= 1) {
      for (int a = tid; a < N; a += 256) {
        int b = a ^ j;
        if (b > a) {
          bool descend = ((a & k) == 0);
          bool sw = descend ? (sk[a] < sk[b]) : (sk[a] > sk[b]);
          if (sw) {
            float tf = sk[a]; sk[a] = sk[b]; sk[b] = tf;
            int ti = si[a]; si[a] = si[b]; si[b] = ti;
          }
        }
      }
      __syncthreads();
    }
  int kc = (s == 0) ? k0 : (s == 1) ? k1 : k2;
  for (int r = tid; r < kc; r += 256) { idxOut[s * N + r] = si[r]; valOut[s * N + r] = sk[r]; }
}

// ---- gather subgraph --------------------------------------------------------
__global__ void gather_nodes(const float* __restrict__ feat, const float* __restrict__ coor,
                             const int* __restrict__ idx, const float* __restrict__ vals,
                             _Float16* __restrict__ hsub, float* __restrict__ hc,
                             float* __restrict__ csub, int n) {
  int t = blockIdx.x * blockDim.x + threadIdx.x;
  if (t < n * 64) {
    int r = t >> 6, d = t & 63;
    float x = feat[idx[r] * 64 + d] * vals[r];
    hc[t] = x; hsub[t] = (_Float16)x;
  }
  if (t < n * 3) csub[t] = coor[idx[t / 3] * 3 + (t % 3)];
}

__global__ void gather_g(const unsigned char* __restrict__ ug2, const int* __restrict__ idx,
                         unsigned char* __restrict__ g, int n, int N) {
  int t = blockIdx.x * blockDim.x + threadIdx.x;
  if (t < n * n) g[t] = ug2[idx[t / n] * N + idx[t % n]];
}

__global__ void scatter_max(const float* __restrict__ hc, const int* __restrict__ idx,
                            float* __restrict__ out, int n) {
  int t = blockIdx.x * blockDim.x + threadIdx.x;
  if (t < n * 64) {
    int r = t >> 6, d = t & 63;
    float v = hc[t];
    float* o = &out[idx[r] * 64 + d];
    if (v > *o) *o = v;
  }
}

// ---------------------------------------------------------------------------
// Edge-MLP kernel: one wave owns one node i, loops over 16-j tiles.
// Chain: e_in(16x160) @We1(160x272) -> silu -> m1(16x288)
//        m1 @We2(288x32) -> silu -> m(16x32), accumulate m_i
//        m @Wc1(32x80) -> silu -> c1(16x96)
//        c1 @Wc2(96x16) -> cw (col 0), accumulate sum_j cw*(c_i - c_j)
// LDS: shared weight fragments (E1|E2|C1|C2 = 56832 halves = 111KB, staged
// once from global) + 8 wave-private activation slices of 9216 halves.
// Total 130560 halves = 255KB of the 320KB WGP LDS.
// ---------------------------------------------------------------------------
#define W_E1 0
#define W_E2 43520
#define W_C1 52736
#define W_C2 55296
#define W_EDGE_HALVES 56832
#define EDGE_WAVE_HALVES 9216
#define EDGE_LDS_HALVES (W_EDGE_HALVES + 8 * EDGE_WAVE_HALVES)

__global__ void __launch_bounds__(256) egnn_edge_kernel(
    const _Float16* __restrict__ hsub, const float* __restrict__ csub,
    const unsigned char* __restrict__ gsub,
    const _Float16* __restrict__ wfL,   // layer base of fragment-packed weights
    const float* __restrict__ be1, const float* __restrict__ be2,
    const float* __restrict__ bc1, const float* __restrict__ bc2,
    float* __restrict__ macc, float* __restrict__ cacc, int n) {
  extern __shared__ _Float16 smem[];
  const int lane = threadIdx.x & 31;
  const int wave = threadIdx.x >> 5;
  const _Float16* wE1 = smem + W_E1;
  const _Float16* wE2 = smem + W_E2;
  const _Float16* wC1 = smem + W_C1;
  const _Float16* wC2 = smem + W_C2;
  _Float16* eIn = smem + W_EDGE_HALVES + wave * EDGE_WAVE_HALVES;
  _Float16* m1 = eIn + 2560;
  _Float16* mm = m1 + 4608;
  _Float16* c1 = mm + 512;

  // cooperative one-time stage of all edge-layer weight fragments into LDS
  {
    const uint4* s = (const uint4*)wfL;
    uint4* d = (uint4*)smem;
    for (int t = threadIdx.x; t < (W_EDGE_HALVES * 2 / 16); t += 256) d[t] = s[t];
  }

  const int i = blockIdx.x * 8 + wave;
  const bool iv = (i < n);
  const int ic = iv ? i : (n - 1);

  // zero K-padding regions once (padded weight rows are also zero => safe)
  for (int r = 0; r < 16; ++r) {
    if (lane < 30) eIn[r * 160 + 130 + lane] = (_Float16)0.f;
    if (lane < 16) { m1[r * 288 + 272 + lane] = (_Float16)0.f; c1[r * 96 + 80 + lane] = (_Float16)0.f; }
  }

  // preload j-loop-invariant biases into per-lane registers (fully-unrolled
  // N-tile loops below keep these arrays in VGPRs, no scratch)
  const int lc = lane & 15;
  float b1r[17], b2r[2], bcr[5];
#pragma unroll
  for (int nt = 0; nt < 17; ++nt) { int col = nt * 16 + lc; b1r[nt] = (col < 260) ? be1[col] : 0.f; }
#pragma unroll
  for (int nt = 0; nt < 2; ++nt) { int col = nt * 16 + lc; b2r[nt] = (col < 17) ? be2[col] : 0.f; }
#pragma unroll
  for (int nt = 0; nt < 5; ++nt) { int col = nt * 16 + lc; bcr[nt] = (col < 68) ? bc1[col] : 0.f; }

  const float cix = csub[ic * 3 + 0], ciy = csub[ic * 3 + 1], ciz = csub[ic * 3 + 2];
  const float bc2v = bc2[0];
  float mP0 = 0.f, mP1 = 0.f;            // per-lane partial m_i (columns)
  float cax = 0.f, cay = 0.f, caz = 0.f; // cw*rel partials (lanes 0 & 16)
  __syncthreads();  // weights staged, pads written

  for (int j0 = 0; j0 < n; j0 += 16) {
    { // stage e_in tile: row r = pair (i, j0+r) = [h_i | h_j | dist | edge]
      const int r = lane & 15;
      const int j = j0 + r;
      const int jc = (j < n) ? j : (n - 1);
      if (lane < 16) {
        const uint4* s = (const uint4*)(hsub + ic * 64);
        uint4* d = (uint4*)(eIn + r * 160);
#pragma unroll
        for (int w = 0; w < 8; ++w) d[w] = s[w];
      } else {
        const uint4* s = (const uint4*)(hsub + jc * 64);
        uint4* d = (uint4*)(eIn + r * 160 + 64);
#pragma unroll
        for (int w = 0; w < 8; ++w) d[w] = s[w];
        const float dx = cix - csub[jc * 3 + 0];
        const float dy = ciy - csub[jc * 3 + 1];
        const float dz = ciz - csub[jc * 3 + 2];
        eIn[r * 160 + 128] = (_Float16)(dx * dx + dy * dy + dz * dz);
        eIn[r * 160 + 129] = (_Float16)(float)gsub[ic * n + jc];
      }
    }
    wave_fence();

    // E1: preload 5 A-fragments (reused across 17 N-tiles), stream B from LDS.
    // Two N-tiles per step -> two independent accumulator chains so the
    // scheduler overlaps one chain's B-loads with the other's WMMAs.
    {
      v16h aE[5];
#pragma unroll
      for (int kt = 0; kt < 5; ++kt) aE[kt] = lds_afrag(eIn, 160, kt);
#pragma unroll
      for (int np = 0; np < 8; ++np) {
        const int nt0 = np * 2, nt1 = np * 2 + 1;
        v8f acc0 = {}, acc1 = {};
#pragma unroll
        for (int kt = 0; kt < 5; ++kt) {
          acc0 = wmma_f32_f16(aE[kt], lds_bfrag(wE1, nt0 * 5 + kt), acc0);
          acc1 = wmma_f32_f16(aE[kt], lds_bfrag(wE1, nt1 * 5 + kt), acc1);
        }
        store_act_silu(m1, 288, nt0, acc0, b1r[nt0]);
        store_act_silu(m1, 288, nt1, acc1, b1r[nt1]);
      }
      { // tail tile nt=16
        v8f acc = {};
#pragma unroll
        for (int kt = 0; kt < 5; ++kt)
          acc = wmma_f32_f16(aE[kt], lds_bfrag(wE1, 16 * 5 + kt), acc);
        store_act_silu(m1, 288, 16, acc, b1r[16]);
      }
    }
    wave_fence();

    // E2: 2 N-tiles x 9 K-slices; accumulate m_i over valid rows
    {
      v16h aM[9];
#pragma unroll
      for (int kt = 0; kt < 9; ++kt) aM[kt] = lds_afrag(m1, 288, kt);
#pragma unroll
      for (int nt = 0; nt < 2; ++nt) {
        v8f acc = {};
#pragma unroll
        for (int kt = 0; kt < 9; ++kt)
          acc = wmma_f32_f16(aM[kt], lds_bfrag(wE2, nt * 9 + kt), acc);
        int col = nt * 16 + lc;
        float bias = b2r[nt];
        int r0 = (lane >> 4) << 3;
        float part = 0.f;
#pragma unroll
        for (int v = 0; v < 8; ++v) {
          float sv = silu_f(acc[v] + bias);
          mm[(r0 + v) * 32 + col] = (_Float16)sv;
          if (j0 + r0 + v < n) part += sv;
        }
        if (nt == 0) mP0 += part; else mP1 += part;
      }
    }
    wave_fence();

    // C1: 5 N-tiles x 1 K-slice (fully unrolled -> 5 independent chains)
    {
      v16h aS = lds_afrag(mm, 32, 0);
#pragma unroll
      for (int nt = 0; nt < 5; ++nt) {
        v8f acc = {};
        acc = wmma_f32_f16(aS, lds_bfrag(wC1, nt), acc);
        store_act_silu(c1, 96, nt, acc, bcr[nt]);
      }
    }
    wave_fence();

    // C2: 1 N-tile x 3 K-slices; column 0 holds cw per row (lanes 0 and 16)
    {
      v8f acc = {};
#pragma unroll
      for (int kt = 0; kt < 3; ++kt)
        acc = wmma_f32_f16(lds_afrag(c1, 96, kt), lds_bfrag(wC2, kt), acc);
      if ((lane & 15) == 0) {
        const int r0 = (lane >> 4) << 3;
#pragma unroll
        for (int v = 0; v < 8; ++v) {
          int j = j0 + r0 + v;
          if (j < n) {
            float cw = acc[v] + bc2v;
            cax += cw * (cix - csub[j * 3 + 0]);
            cay += cw * (ciy - csub[j * 3 + 1]);
            caz += cw * (ciz - csub[j * 3 + 2]);
          }
        }
      }
    }
    wave_fence();
  }

  // reduce wave32 halves (rows 0-7 in lanes 0-15, rows 8-15 in lanes 16-31)
  mP0 += __shfl_xor(mP0, 16);
  mP1 += __shfl_xor(mP1, 16);
  cax += __shfl_xor(cax, 16);
  cay += __shfl_xor(cay, 16);
  caz += __shfl_xor(caz, 16);
  if (iv) {
    if (lane < 16) macc[i * 17 + lane] = mP0;
    if (lane == 0) {
      macc[i * 17 + 16] = mP1;
      cacc[i * 3 + 0] = cax; cacc[i * 3 + 1] = cay; cacc[i * 3 + 2] = caz;
    }
  }
}

// ---------------------------------------------------------------------------
// Node-MLP kernel: tiles of 16 nodes. cat=[h|m_i] (16x96) @Wn1(96x128) -> silu
// -> (16x128) @Wn2(128x64); hc = relu(2*hc + mlp); also c += cacc.
// ---------------------------------------------------------------------------
__global__ void __launch_bounds__(128) egnn_node_kernel(
    const _Float16* __restrict__ Wn1f, const _Float16* __restrict__ Wn2f,
    const float* __restrict__ bn1, const float* __restrict__ bn2,
    float* __restrict__ hc, _Float16* __restrict__ hsub,
    const float* __restrict__ macc, float* __restrict__ csub,
    const float* __restrict__ cacc, int n) {
  __shared__ _Float16 smem[4 * (1536 + 2048)];
  const int lane = threadIdx.x & 31;
  const int wave = threadIdx.x >> 5;
  _Float16* cat = smem + wave * (1536 + 2048);
  _Float16* m1n = cat + 1536;
  const int ntiles = (n + 15) >> 4;
  const int tile = blockIdx.x * 4 + wave;
  const bool tv = (tile < ntiles);
  const int tb = (tv ? tile : 0) * 16;
  const int lc = lane & 15;

  float bn1r[8], bn2r[4];
#pragma unroll
  for (int nt = 0; nt < 8; ++nt) bn1r[nt] = bn1[nt * 16 + lc];
#pragma unroll
  for (int nt = 0; nt < 4; ++nt) bn2r[nt] = bn2[nt * 16 + lc];

  for (int t = lane; t < 16 * 96; t += 32) {
    int r = t / 96, col = t % 96;
    int ii = tb + r; if (ii >= n) ii = n - 1;
    float x = (col < 64) ? hc[ii * 64 + col]
            : ((col < 81) ? macc[ii * 17 + (col - 64)] : 0.f);
    cat[t] = (_Float16)x;
  }
  wave_fence();
  {
    v16h aC[3];
#pragma unroll
    for (int kt = 0; kt < 3; ++kt) aC[kt] = lds_afrag(cat, 96, kt);
#pragma unroll
    for (int nt = 0; nt < 8; ++nt) {
      v8f acc = {};
#pragma unroll
      for (int kt = 0; kt < 3; ++kt)
        acc = wmma_f32_f16(aC[kt], *(const v16h*)(Wn1f + ((nt * 3 + kt) * 32 + lane) * 16), acc);
      store_act_silu(m1n, 128, nt, acc, bn1r[nt]);
    }
  }
  wave_fence();
  {
    v16h aM[4];
#pragma unroll
    for (int kt = 0; kt < 4; ++kt) aM[kt] = lds_afrag(m1n, 128, kt);
#pragma unroll
    for (int nt = 0; nt < 4; ++nt) {
      v8f acc = {};
#pragma unroll
      for (int kt = 0; kt < 4; ++kt)
        acc = wmma_f32_f16(aM[kt], *(const v16h*)(Wn2f + ((nt * 4 + kt) * 32 + lane) * 16), acc);
      const int col = nt * 16 + lc;
      const float bias = bn2r[nt];
      const int r0 = (lane >> 4) << 3;
#pragma unroll
      for (int v = 0; v < 8; ++v) {
        int ii = tb + r0 + v;
        if (tv && ii < n) {
          float hin = hc[ii * 64 + col];
          float hn = 2.f * hin + acc[v] + bias;  // hc + (hc + mlp)
          hn = hn > 0.f ? hn : 0.f;              // relu
          hc[ii * 64 + col] = hn;
          hsub[ii * 64 + col] = (_Float16)hn;
        }
      }
    }
  }
  __syncthreads();
  for (int t = threadIdx.x; t < 64 * 3; t += 128) {
    int ii = blockIdx.x * 64 + t / 3;
    if (ii < n) csub[ii * 3 + (t % 3)] += cacc[ii * 3 + (t % 3)];
  }
}

// ---------------------------------------------------------------------------
extern "C" void kernel_launch(void* const* d_in, const int* in_sizes, int n_in,
                              void* d_out, int out_size, void* d_ws, size_t ws_size,
                              hipStream_t stream) {
  const float* feat = (const float*)d_in[0];
  const float* coor = (const float*)d_in[1];
  const float* edge = (const float*)d_in[2];
  const float* We1 = (const float*)d_in[3];
  const float* be1 = (const float*)d_in[4];
  const float* We2 = (const float*)d_in[5];
  const float* be2 = (const float*)d_in[6];
  const float* Wc1 = (const float*)d_in[7];
  const float* bc1 = (const float*)d_in[8];
  const float* Wc2 = (const float*)d_in[9];
  const float* bc2 = (const float*)d_in[10];
  const float* Wn1 = (const float*)d_in[11];
  const float* bn1 = (const float*)d_in[12];
  const float* Wn2 = (const float*)d_in[13];
  const float* bn2 = (const float*)d_in[14];
  const float* wp = (const float*)d_in[15];
  const float* bp = (const float*)d_in[16];
  float* out = (float*)d_out;

  const int N = in_sizes[0] / 64;  // 512
  const int words = N / 32;
  int kcs[3];
  const double KS[3] = {0.9, 0.7, 0.6};
  for (int s = 0; s < 3; ++s) {
    int k = (int)(KS[s] * N);
    kcs[s] = k < 2 ? 2 : k;
  }
  const int n0 = kcs[0];

  char* ws = (char*)d_ws;
  size_t off = 0;
  auto take = [&](size_t bytes) -> char* {
    char* p = ws + off;
    off = (off + bytes + 255) & ~(size_t)255;
    return p;
  };
  unsigned char* ug2 = (unsigned char*)take((size_t)N * N);
  unsigned* rowm = (unsigned*)take((size_t)N * words * 4);
  unsigned* colm = (unsigned*)take((size_t)N * words * 4);
  int* idxb = (int*)take((size_t)3 * N * 4);
  float* valb = (float*)take((size_t)3 * N * 4);
  _Float16* hsub = (_Float16*)take((size_t)n0 * 64 * 2);
  float* hc = (float*)take((size_t)n0 * 64 * 4);
  float* csub = (float*)take((size_t)n0 * 3 * 4);
  unsigned char* gsub = (unsigned char*)take((size_t)n0 * n0);
  float* macc = (float*)take((size_t)n0 * 17 * 4);
  float* cacc = (float*)take((size_t)n0 * 3 * 4);
  const size_t LS = 77312;  // halves per layer of fragment-packed weights
  _Float16* wf = (_Float16*)take((size_t)3 * LS * 2);
  (void)ws_size; (void)n_in; (void)out_size;

  const int TPB = 256;
  // fragment offsets within a layer (halves); E1|E2|C1|C2 contiguous = edge block
  const size_t oN1 = 56832, oN2 = 69120;

  for (int l = 0; l < 3; ++l) {
    _Float16* base = wf + (size_t)l * LS;
    prep_w<<<(5 * 17 * 512 + TPB - 1) / TPB, TPB, 0, stream>>>(We1 + l * 130 * 260, base + W_E1, 130, 260, 5, 17);
    prep_w<<<(9 * 2 * 512 + TPB - 1) / TPB, TPB, 0, stream>>>(We2 + l * 260 * 17, base + W_E2, 260, 17, 9, 2);
    prep_w<<<(1 * 5 * 512 + TPB - 1) / TPB, TPB, 0, stream>>>(Wc1 + l * 17 * 68, base + W_C1, 17, 68, 1, 5);
    prep_w<<<(3 * 1 * 512 + TPB - 1) / TPB, TPB, 0, stream>>>(Wc2 + l * 68 * 1, base + W_C2, 68, 1, 3, 1);
    prep_w<<<(3 * 8 * 512 + TPB - 1) / TPB, TPB, 0, stream>>>(Wn1 + l * 81 * 128, base + oN1, 81, 128, 3, 8);
    prep_w<<<(4 * 4 * 512 + TPB - 1) / TPB, TPB, 0, stream>>>(Wn2 + l * 128 * 64, base + oN2, 128, 64, 4, 4);
  }
  zero_f32<<<(N * 64 + TPB - 1) / TPB, TPB, 0, stream>>>(out, N * 64);
  pack_masks<<<(N * words + TPB - 1) / TPB, TPB, 0, stream>>>(edge, rowm, colm, N);
  ug2_kernel<<<(N * N + TPB - 1) / TPB, TPB, 0, stream>>>(rowm, colm, ug2, N);
  topk_kernel<<<3, 256, 0, stream>>>(feat, wp, bp, idxb, valb, N, kcs[0], kcs[1], kcs[2]);

  for (int s = 0; s < 3; ++s) {
    const int n = kcs[s];
    const int* idx = idxb + s * N;
    const float* vals = valb + s * N;
    gather_nodes<<<(n * 64 + TPB - 1) / TPB, TPB, 0, stream>>>(feat, coor, idx, vals, hsub, hc, csub, n);
    gather_g<<<(n * n + TPB - 1) / TPB, TPB, 0, stream>>>(ug2, idx, gsub, n, N);
    for (int l = 0; l < 3; ++l) {
      const _Float16* base = wf + (size_t)l * LS;
      const size_t shmem = EDGE_LDS_HALVES * sizeof(_Float16);  // ~255 KB of the 320 KB WGP LDS
      egnn_edge_kernel<<<(n + 7) / 8, 256, shmem, stream>>>(
          hsub, csub, gsub, base,
          be1 + l * 260, be2 + l * 17, bc1 + l * 68, bc2 + l, macc, cacc, n);
      const int ntiles = (n + 15) / 16;
      egnn_node_kernel<<<(ntiles + 3) / 4, 128, 0, stream>>>(
          base + oN1, base + oN2, bn1 + l * 128, bn2 + l * 64, hc, hsub, macc, csub, cacc, n);
    }
    scatter_max<<<(n * 64 + TPB - 1) / TPB, TPB, 0, stream>>>(hc, idx, out, n);
  }
}
